// MultiHeadAttention_82094004895779
// MI455X (gfx1250) — matrix operand order repaired and checked by reading, not compiled, not run on hardware
//
#include <hip/hip_runtime.h>
#include <hip/hip_bf16.h>

// Problem constants (B=1)
#define S_LEN    4096
#define D_DIM    1024
#define NHEADS   16
#define HEAD_DIM 64

typedef __attribute__((ext_vector_type(16))) __bf16   bf16x16;
typedef __attribute__((ext_vector_type(8)))  __bf16   bf16x8;
typedef __attribute__((ext_vector_type(2)))  __bf16   bf16x2;
typedef __attribute__((ext_vector_type(8)))  float    f32x8;
typedef __attribute__((ext_vector_type(4)))  unsigned u32x4;
typedef __attribute__((ext_vector_type(8)))  int      i32x8;
typedef __attribute__((ext_vector_type(4)))  int      i32x4;

// ---------------------------------------------------------------------------
// Fragment loaders (wave32); this file's contiguous k order, same for A and B.
// A (16x32 bf16): lane L (row=L%16, h=L/16): elem j = A[row][k0+h*16+j]
// B (32x16 bf16): lane L (col=L%16, h=L/16): elem j = B[h*16+j][col] ->
//                 one contiguous 16-element run when "columns" are rows of a
//                 row-major matrix (X @ W^T pattern).
// ---------------------------------------------------------------------------
__device__ __forceinline__ bf16x16 load_fragA(const __bf16* __restrict__ base,
                                              int ld, int row0, int k0, int lane) {
  const int r = row0 + (lane & 15);
  const int h = lane >> 4;
  const __bf16* p = base + (size_t)r * ld + k0 + h * 16;
  bf16x8 lo = *(const bf16x8*)p;
  bf16x8 hi = *(const bf16x8*)(p + 8);
  bf16x16 out;
#pragma unroll
  for (int i = 0; i < 8; ++i) { out[i] = lo[i]; out[i + 8] = hi[i]; }
  return out;
}

__device__ __forceinline__ bf16x16 load_fragB(const __bf16* p) {
  return *(const bf16x16*)p;
}

__device__ __forceinline__ f32x8 wmma_bf16(const bf16x16& a, const bf16x16& b, const f32x8& c) {
  return __builtin_amdgcn_wmma_f32_16x16x32_bf16(false, a, false, b, (short)0, c, false, false);
}

// ---------------------------------------------------------------------------
// DPP butterfly max within 16-lane rows (C/D-layout rows live across 16 lanes).
// xor1/2/4 via DPP8 in-byte-group permutes; xor8 == row_ror:8 inside a 16-lane
// DPP row. Pure VALU -> no LDS (ds_bpermute) traffic.
// ---------------------------------------------------------------------------
#define SEL8(a,b,c,d,e,f,g,h) \
  ((a)|((b)<<3)|((c)<<6)|((d)<<9)|((e)<<12)|((f)<<15)|((g)<<18)|((h)<<21))

__device__ __forceinline__ float row_max16(float x) {
  x = fmaxf(x, __int_as_float(__builtin_amdgcn_mov_dpp8(__float_as_int(x),
            SEL8(1,0,3,2,5,4,7,6))));                        // xor 1
  x = fmaxf(x, __int_as_float(__builtin_amdgcn_mov_dpp8(__float_as_int(x),
            SEL8(2,3,0,1,6,7,4,5))));                        // xor 2
  x = fmaxf(x, __int_as_float(__builtin_amdgcn_mov_dpp8(__float_as_int(x),
            SEL8(4,5,6,7,0,1,2,3))));                        // xor 4
  x = fmaxf(x, __int_as_float(__builtin_amdgcn_mov_dpp(__float_as_int(x),
            0x128 /*row_ror:8*/, 0xf, 0xf, false)));         // xor 8 (in row16)
  return x;
}

// ---------------------------------------------------------------------------
// Tensor Data Mover: 2D bf16 tile (tile_d0 x tile_d1 elems) global -> LDS.
// Descriptor per CDNA5 ISA ch.8 (D# group0/group1). LDS byte address = low 32
// bits of the generic pointer (LDS aperture, ISA 10.2). Arity differs between
// ROCm clang-22 (5 args) and clang-23 (6 args).
// ---------------------------------------------------------------------------
__device__ __forceinline__ void tdm_load_2d_bf16(const void* lds_dst,
                                                 const void* gsrc,
                                                 unsigned tile_d0, unsigned tile_d1,
                                                 unsigned tensor_d0,
                                                 unsigned stride_elems) {
  const unsigned long long ga = (unsigned long long)gsrc;
  const unsigned lds_off = (unsigned)(unsigned long long)lds_dst;
  const unsigned tensor_d1 = 1u << 20;

  u32x4 g0;
  g0[0] = 1u;                                           // count=1, user mode
  g0[1] = lds_off;                                      // lds_addr
  g0[2] = (unsigned)ga;                                 // global_addr[31:0]
  g0[3] = (unsigned)((ga >> 32) & 0x01ffffffull) | (2u << 30);  // addr hi | type=2

  i32x8 g1;
  g1[0] = 0x10000;                                      // data_size=1 (2 bytes)
  g1[1] = (int)((tensor_d0 & 0xffffu) << 16);           // tensor_dim0 lo16
  g1[2] = (int)((tensor_d0 >> 16) | ((tensor_d1 & 0xffffu) << 16));
  g1[3] = (int)((tensor_d1 >> 16) | (tile_d0 << 16));   // | tile_dim0
  g1[4] = (int)(tile_d1 & 0xffffu);                     // tile_dim1 (tile_dim2=0)
  g1[5] = (int)stride_elems;                            // tensor_dim0_stride lo32
  g1[6] = 0;                                            // stride hi16 | dim1_stride
  g1[7] = 0;

  i32x4 gz; gz[0] = 0; gz[1] = 0; gz[2] = 0; gz[3] = 0;
#if defined(__clang_major__) && (__clang_major__ >= 23)
  i32x8 gz8;
#pragma unroll
  for (int i = 0; i < 8; ++i) gz8[i] = 0;
  __builtin_amdgcn_tensor_load_to_lds(g0, g1, gz, gz, gz8, 0);
#else
  __builtin_amdgcn_tensor_load_to_lds(g0, g1, gz, gz, 0);
#endif
}

// ---------------------------------------------------------------------------
// fp32 -> bf16 conversion
// ---------------------------------------------------------------------------
__global__ void cvt_f32_bf16_kernel(const float* __restrict__ in,
                                    __bf16* __restrict__ out, int n) {
  int i = blockIdx.x * blockDim.x + threadIdx.x;
  if (i < n) out[i] = (__bf16)in[i];
}

// ---------------------------------------------------------------------------
// GEMM: C[4096][1024] = A(bf16 MxK) @ W^T (W bf16 row-major NxK), K=1024.
// Block = 8 waves; block tile 128(M) x 64(N); wave tile 16x64.
// W k-slab (64 rows x 32 k) streamed into double-buffered LDS by the TDM
// (wave 0 issues, s_wait_tensorcnt, one barrier per k-step).
// MODE 0: fused RoPE epilogue -> bf16 [head][pos][64]   (Q, K)
// MODE 1: plain bf16 epilogue -> [head][pos][64]        (V)
// MODE 2: fp32 epilogue, row-major [pos][1024]          (final O-proj)
// ---------------------------------------------------------------------------
template <int MODE>
__global__ void __launch_bounds__(256)
gemm_wmma_kernel(const __bf16* __restrict__ A,
                 const __bf16* __restrict__ W,
                 const float* __restrict__ cosp,
                 const float* __restrict__ sinp,
                 __bf16* __restrict__ outb,
                 float* __restrict__ outf) {
  __shared__ __bf16 Wtile[2][64 * 32];

  const int tid  = threadIdx.x;
  const int lane = tid & 31;
  const int wave = tid >> 5;
  const int n16  = lane & 15;
  const int h    = lane >> 4;
  const int Mtile = blockIdx.x * 128 + wave * 16;
  const int Nbase = blockIdx.y * 64;

  f32x8 acc[4];
#pragma unroll
  for (int f = 0; f < 4; ++f)
#pragma unroll
    for (int j = 0; j < 8; ++j) acc[f][j] = 0.0f;

  const __bf16* Wblk = W + (size_t)Nbase * D_DIM;   // 64 rows x 1024

  if (wave == 0) {
    tdm_load_2d_bf16(&Wtile[0][0], Wblk, /*tile_d0*/32, /*tile_d1*/64,
                     /*tensor_d0*/D_DIM, /*stride*/D_DIM);
    __builtin_amdgcn_s_wait_tensorcnt(0);
  }
  __syncthreads();

  int buf = 0;
  for (int kk = 0; kk < D_DIM; kk += 32) {
    if (wave == 0 && (kk + 32) < D_DIM) {
      tdm_load_2d_bf16(&Wtile[buf ^ 1][0], Wblk + kk + 32, 32, 64, D_DIM, D_DIM);
    }

    bf16x16 a = load_fragA(A, D_DIM, Mtile, kk, lane);
#pragma unroll
    for (int f = 0; f < 4; ++f) {
      bf16x16 b = load_fragB(&Wtile[buf][(f * 16 + n16) * 32 + h * 16]);
      acc[f] = wmma_bf16(a, b, acc[f]);
    }

    if (wave == 0) __builtin_amdgcn_s_wait_tensorcnt(0);
    __syncthreads();
    buf ^= 1;
  }

  if (MODE == 0) {
    // RoPE: one head per N-tile (64 cols). x1 = cols 0..31, x2 = cols 32..63.
    const int head = blockIdx.y;
    __bf16* dst = outb + (size_t)head * S_LEN * HEAD_DIM;
#pragma unroll
    for (int v = 0; v < 8; ++v) {
      const int pos = Mtile + v + 8 * h;
#pragma unroll
      for (int f = 0; f < 2; ++f) {
        const int i = f * 16 + n16;                 // 0..31
        const float c = cosp[pos * 32 + i];
        const float s = sinp[pos * 32 + i];
        const float x1 = acc[f][v];
        const float x2 = acc[f + 2][v];
        dst[(size_t)pos * HEAD_DIM + i]      = (__bf16)(x1 * c - x2 * s);
        dst[(size_t)pos * HEAD_DIM + i + 32] = (__bf16)(x2 * c + x1 * s);
      }
    }
  } else if (MODE == 1) {
    const int head = blockIdx.y;
    __bf16* dst = outb + (size_t)head * S_LEN * HEAD_DIM;
#pragma unroll
    for (int v = 0; v < 8; ++v) {
      const int pos = Mtile + v + 8 * h;
#pragma unroll
      for (int f = 0; f < 4; ++f)
        dst[(size_t)pos * HEAD_DIM + f * 16 + n16] = (__bf16)acc[f][v];
    }
  } else {
#pragma unroll
    for (int v = 0; v < 8; ++v) {
      const int pos = Mtile + v + 8 * h;
#pragma unroll
      for (int f = 0; f < 4; ++f)
        outf[(size_t)pos * D_DIM + Nbase + f * 16 + n16] = acc[f][v];
    }
  }
}

// ---------------------------------------------------------------------------
// Flash attention. Block = 8 waves = (head = blockIdx.y, 128 q-rows);
// each wave owns 16 q-rows, online softmax over keys in tiles of 32.
// Key index inside a tile uses permuted order k' = 2*(key%16) + key/16 so the
// (p0,p1) pair of each lane packs into one ds_store_b32; the V tile is staged
// with the same permutation (WMMA contraction is order-invariant).
// Row-max: DPP butterfly. Row-sum: WMMA against an all-ones B fragment.
// ---------------------------------------------------------------------------
__global__ void __launch_bounds__(256)
flash_attn_kernel(const __bf16* __restrict__ Q,
                  const __bf16* __restrict__ K,
                  const __bf16* __restrict__ V,
                  const float* __restrict__ mask,
                  __bf16* __restrict__ out) {
  __shared__ __bf16 Ktile[32 * 64];
  __shared__ __bf16 Vtile[64 * 32];
  __shared__ __bf16 Ptile[8][16 * 32];

  const int tid  = threadIdx.x;
  const int lane = tid & 31;
  const int wave = tid >> 5;
  const int n16  = lane & 15;
  const int h    = lane >> 4;
  const int head = blockIdx.y;
  const int qbase = blockIdx.x * 128 + wave * 16;

  const __bf16* Qh = Q + (size_t)head * S_LEN * HEAD_DIM;
  const __bf16* Kh = K + (size_t)head * S_LEN * HEAD_DIM;
  const __bf16* Vh = V + (size_t)head * S_LEN * HEAD_DIM;

  bf16x16 qa[2];
  qa[0] = load_fragA(Qh, HEAD_DIM, qbase, 0, lane);
  qa[1] = load_fragA(Qh, HEAD_DIM, qbase, 32, lane);

  bf16x16 ones;
#pragma unroll
  for (int i = 0; i < 16; ++i) ones[i] = (__bf16)1.0f;

  f32x8 o[4];
#pragma unroll
  for (int f = 0; f < 4; ++f)
#pragma unroll
    for (int j = 0; j < 8; ++j) o[f][j] = 0.0f;

  float m_run[8], l_run[8], alpha[8];
#pragma unroll
  for (int v = 0; v < 8; ++v) { m_run[v] = -1e30f; l_run[v] = 0.0f; }

  const int skey = tid >> 3;                      // 0..31 : staged key row
  const int scol = (tid & 7) * 8;                 // 0,8,..,56 : hd column base
  const int pkey = ((skey & 15) << 1) | (skey >> 4);  // permuted key slot

  for (int kb = 0; kb < S_LEN; kb += 32) {
    __syncthreads();
    {
      const __bf16* kg = Kh + (size_t)(kb + skey) * HEAD_DIM + scol;
      const __bf16* vg = Vh + (size_t)(kb + skey) * HEAD_DIM + scol;
      bf16x8 kv = *(const bf16x8*)kg;
      *(bf16x8*)(&Ktile[skey * 64 + scol]) = kv;
      bf16x8 vv = *(const bf16x8*)vg;
#pragma unroll
      for (int i = 0; i < 8; ++i) Vtile[(scol + i) * 32 + pkey] = vv[i];
      if (kb + 32 < S_LEN) {
        __builtin_prefetch(kg + 32 * HEAD_DIM, 0, 1);
        __builtin_prefetch(vg + 32 * HEAD_DIM, 0, 1);
      }
    }
    __syncthreads();

    // ---- scores S = (Q @ K^T) * scale + mask, 16 rows x 32 keys ----
    f32x8 s0, s1;
#pragma unroll
    for (int j = 0; j < 8; ++j) { s0[j] = 0.0f; s1[j] = 0.0f; }
#pragma unroll
    for (int s = 0; s < 2; ++s) {
      bf16x16 b0 = load_fragB(&Ktile[(n16) * 64 + s * 32 + h * 16]);
      bf16x16 b1 = load_fragB(&Ktile[(16 + n16) * 64 + s * 32 + h * 16]);
      s0 = wmma_bf16(qa[s], b0, s0);
      s1 = wmma_bf16(qa[s], b1, s1);
    }
    const float add0 = (1.0f - mask[kb + n16]) * -10000.0f;
    const float add1 = (1.0f - mask[kb + 16 + n16]) * -10000.0f;
#pragma unroll
    for (int v = 0; v < 8; ++v) {
      s0[v] = s0[v] * 0.125f + add0;   // 1/sqrt(64)
      s1[v] = s1[v] * 0.125f + add1;
    }

    // ---- online softmax: DPP row-max, exp, packed P store ----
#pragma unroll
    for (int v = 0; v < 8; ++v) {
      const float t = row_max16(fmaxf(s0[v], s1[v]));
      const float m_new = fmaxf(m_run[v], t);
      alpha[v] = __expf(m_run[v] - m_new);
      m_run[v] = m_new;
      const float p0 = __expf(s0[v] - m_new);
      const float p1 = __expf(s1[v] - m_new);
      const int row = v + 8 * h;                 // C-layout row for this VGPR
      bf16x2 pp;
      pp[0] = (__bf16)p0;
      pp[1] = (__bf16)p1;
      *(bf16x2*)(&Ptile[wave][row * 32 + 2 * n16]) = pp;
    }
#pragma unroll
    for (int f = 0; f < 4; ++f)
#pragma unroll
      for (int v = 0; v < 8; ++v) o[f][v] *= alpha[v];

    // ---- P (16x32, A-fragment layout) from per-wave LDS ----
    bf16x16 pa;
    {
      const __bf16* p = &Ptile[wave][n16 * 32 + h * 16];
      bf16x8 lo = *(const bf16x8*)p;
      bf16x8 hi = *(const bf16x8*)(p + 8);
#pragma unroll
      for (int i = 0; i < 8; ++i) { pa[i] = lo[i]; pa[i + 8] = hi[i]; }
    }

    // ---- row-sum via WMMA against ones; l update ----
    f32x8 rs;
#pragma unroll
    for (int j = 0; j < 8; ++j) rs[j] = 0.0f;
    rs = wmma_bf16(pa, ones, rs);
#pragma unroll
    for (int v = 0; v < 8; ++v) l_run[v] = l_run[v] * alpha[v] + rs[v];

    // ---- O += P @ V ----
#pragma unroll
    for (int f = 0; f < 4; ++f) {
      bf16x16 bv = load_fragB(&Vtile[(f * 16 + n16) * 32 + h * 16]);
      o[f] = wmma_bf16(pa, bv, o[f]);
    }
  }

  // ---- normalize and store bf16 attn output, row-major [pos][1024] ----
#pragma unroll
  for (int v = 0; v < 8; ++v) {
    const int pos = qbase + v + 8 * h;
    const float inv = 1.0f / l_run[v];
#pragma unroll
    for (int f = 0; f < 4; ++f)
      out[(size_t)pos * D_DIM + head * HEAD_DIM + f * 16 + n16] =
          (__bf16)(o[f][v] * inv);
  }
}

// ---------------------------------------------------------------------------
// Launch
// ---------------------------------------------------------------------------
extern "C" void kernel_launch(void* const* d_in, const int* in_sizes, int n_in,
                              void* d_out, int out_size, void* d_ws, size_t ws_size,
                              hipStream_t stream) {
  const float* hs   = (const float*)d_in[0];
  const float* cosp = (const float*)d_in[1];
  const float* sinp = (const float*)d_in[2];
  const float* mask = (const float*)d_in[3];
  const float* Wq   = (const float*)d_in[4];
  const float* Wk   = (const float*)d_in[5];
  const float* Wv   = (const float*)d_in[6];
  const float* Wo   = (const float*)d_in[7];

  const int NX = S_LEN * D_DIM;   // 4,194,304
  const int NW = D_DIM * D_DIM;   // 1,048,576

  __bf16* Xb  = (__bf16*)d_ws;
  __bf16* Wqb = Xb  + NX;
  __bf16* Wkb = Wqb + NW;
  __bf16* Wvb = Wkb + NW;
  __bf16* Wob = Wvb + NW;
  __bf16* Qb  = Wob + NW;
  __bf16* Kb  = Qb  + NX;
  __bf16* Vb  = Kb  + NX;
  __bf16* Ab  = Vb  + NX;        // total 48 MB of workspace

  const int thr = 256;
  cvt_f32_bf16_kernel<<<(NX + thr - 1) / thr, thr, 0, stream>>>(hs, Xb, NX);
  cvt_f32_bf16_kernel<<<(NW + thr - 1) / thr, thr, 0, stream>>>(Wq, Wqb, NW);
  cvt_f32_bf16_kernel<<<(NW + thr - 1) / thr, thr, 0, stream>>>(Wk, Wkb, NW);
  cvt_f32_bf16_kernel<<<(NW + thr - 1) / thr, thr, 0, stream>>>(Wv, Wvb, NW);
  cvt_f32_bf16_kernel<<<(NW + thr - 1) / thr, thr, 0, stream>>>(Wo, Wob, NW);

  dim3 grid(S_LEN / 128, NHEADS);   // 32 x 16
  dim3 block(256);
  gemm_wmma_kernel<0><<<grid, block, 0, stream>>>(Xb, Wqb, cosp, sinp, Qb, nullptr);
  gemm_wmma_kernel<0><<<grid, block, 0, stream>>>(Xb, Wkb, cosp, sinp, Kb, nullptr);
  gemm_wmma_kernel<1><<<grid, block, 0, stream>>>(Xb, Wvb, nullptr, nullptr, Vb, nullptr);

  flash_attn_kernel<<<grid, block, 0, stream>>>(Qb, Kb, Vb, mask, Ab);

  gemm_wmma_kernel<2><<<grid, block, 0, stream>>>(Ab, Wob, nullptr, nullptr, nullptr,
                                                  (float*)d_out);
}
